// CrfRnnLayer_89670327206680
// MI455X (gfx1250) — matrix operand order repaired and checked by reading, not compiled, not run on hardware
//
#include <hip/hip_runtime.h>
#include <hip/hip_bf16.h>

typedef __attribute__((ext_vector_type(16))) _Float16 v16h;
typedef __attribute__((ext_vector_type(8)))  float    v8f;

#define N_PIX   6400
#define WDIM    80
#define NCLS    21
#define CPAD    32
#define NITER   5
#define JCHUNK  32
#define NCHUNK  (N_PIX / JCHUNK)          // 200
#define WPB     8                         // waves per block
#define CPW     (NCHUNK / WPB)            // 25 chunks per wave

// sqrt(0.5 * log2(e)): pre-scaling features turns exp(-0.5*d^2) into exp2(-d'^2),
// so the whole score is FMA-chain + one v_exp_f32 with a free neg source modifier.
#define FSCALE  0.84932180028801904272f

// ---------------- feature kernel: feat[i][8] = {fs0,fs1, fb0..fb4, 0} (pre-scaled) -------
__global__ void crf_feat_kernel(const float* __restrict__ image,
                                float* __restrict__ feat) {
    int i = blockIdx.x * blockDim.x + threadIdx.x;
    if (i >= N_PIX) return;
    float y = (float)(i / WDIM), x = (float)(i % WDIM);
    const float* px = image + (size_t)i * 3;
    float* f = feat + (size_t)i * 8;
    f[0] = y * (FSCALE / 3.0f);      // theta_gamma
    f[1] = x * (FSCALE / 3.0f);
    f[2] = y * (FSCALE / 160.0f);    // theta_alpha
    f[3] = x * (FSCALE / 160.0f);
    f[4] = px[0] * (FSCALE / 3.0f);  // theta_beta
    f[5] = px[1] * (FSCALE / 3.0f);
    f[6] = px[2] * (FSCALE / 3.0f);
    f[7] = 0.0f;
}

// ------------- fold spatial_w/bilateral_w through compat: E[c'][c] = (compat@W)[c][c'] ---
__global__ void crf_mix_kernel(const float* __restrict__ sw, const float* __restrict__ bw,
                               const float* __restrict__ cm,
                               float* __restrict__ es, float* __restrict__ eb) {
    int idx = blockIdx.x * blockDim.x + threadIdx.x;
    if (idx >= NCLS * NCLS) return;
    int cp = idx / NCLS, c = idx % NCLS;
    float s = 0.f, b = 0.f;
    for (int k = 0; k < NCLS; ++k) {
        float ck = cm[c * NCLS + k];
        s += ck * sw[k * NCLS + cp];
        b += ck * bw[k * NCLS + cp];
    }
    es[cp * NCLS + c] = s;
    eb[cp * NCLS + c] = b;
}

// ------------- softmax + write p TRANSPOSED fp16, row 21 = ones, rows 22..31 = 0 ---------
__global__ void crf_softmax_kernel(const float* __restrict__ q,
                                   _Float16* __restrict__ pT) {
    int i = blockIdx.x * blockDim.x + threadIdx.x;
    if (i >= N_PIX) return;
    const float* qi = q + (size_t)i * NCLS;
    float m = qi[0];
#pragma unroll
    for (int c = 1; c < NCLS; ++c) m = fmaxf(m, qi[c]);
    float e[NCLS]; float s = 0.f;
#pragma unroll
    for (int c = 0; c < NCLS; ++c) { e[c] = __expf(qi[c] - m); s += e[c]; }
    float inv = 1.0f / s;
#pragma unroll
    for (int c = 0; c < NCLS; ++c) pT[(size_t)c * N_PIX + i] = (_Float16)(e[c] * inv);
    pT[(size_t)21 * N_PIX + i] = (_Float16)1.0f;   // ones column -> norm via same WMMA
#pragma unroll
    for (int c = 22; c < CPAD; ++c) pT[(size_t)c * N_PIX + i] = (_Float16)0.0f;
}

// ------------- fused tile kernel: scores (VALU+exp2) -> fp16 WMMA -> normalize+mix -------
__global__ __launch_bounds__(WPB * 32, 1)
void crf_tile_kernel(const float* __restrict__ feat, const _Float16* __restrict__ pT,
                     const float* __restrict__ u, const float* __restrict__ es,
                     const float* __restrict__ eb, float* __restrict__ qout) {
    __shared__ __attribute__((aligned(16))) float fl[WPB][JCHUNK][8]; // staged j-features
    __shared__ float MsL[WPB][16][CPAD];
    __shared__ float MbL[WPB][16][CPAD];
    __shared__ float MsR[16][24];
    __shared__ float MbR[16][24];

    const int t    = threadIdx.x;
    const int w    = t >> 5;
    const int lane = t & 31;
    const int m16  = lane & 15;
    const bool hi  = lane >= 16;
    const int i0   = blockIdx.x * 16;

    // this lane's output row features (A-matrix: lane L holds row M = L%16)
    const float* fi = feat + (size_t)(i0 + m16) * 8;
    const float fs0 = fi[0], fs1 = fi[1];
    const float fb0 = fi[2], fb1 = fi[3], fb2 = fi[4], fb3 = fi[5], fb4 = fi[6];

    const int khalf = hi ? 8 : 0;    // A-layout K sets {0..7,16..23} / {8..15,24..31}
    const int jhalf = hi ? 16 : 0;   // contiguous K halves {0..15} / {16..31}, used for both A and B

    v8f accs0 = {0,0,0,0,0,0,0,0}, accs1 = {0,0,0,0,0,0,0,0};
    v8f accb0 = {0,0,0,0,0,0,0,0}, accb1 = {0,0,0,0,0,0,0,0};

    for (int cc = 0; cc < CPW; ++cc) {
        const int j0 = (w * CPW + cc) * JCHUNK;

        // stage 32 j-feature rows into wave-private LDS (1 row per lane, 2x b128)
        const float4* fg = (const float4*)(feat + (size_t)(j0 + lane) * 8);
        *((float4*)&fl[w][lane][0]) = fg[0];
        *((float4*)&fl[w][lane][4]) = fg[1];

        // build fp16 A matrices (Gaussian scores) directly in WMMA A layout
        v16h As, Ab;
#pragma unroll
        for (int e = 0; e < 16; ++e) {
            const int k = e + jhalf;
            const float* fj = &fl[w][k][0];
            float d0 = fs0 - fj[0], d1 = fs1 - fj[1];
            float ds = d0 * d0 + d1 * d1;
            float b0 = fb0 - fj[2], b1 = fb1 - fj[3], b2 = fb2 - fj[4],
                  b3 = fb3 - fj[5], b4 = fb4 - fj[6];
            float db = b0*b0 + b1*b1 + b2*b2 + b3*b3 + b4*b4;
            // features pre-scaled by sqrt(0.5*log2 e): exp(-0.5 d^2) == exp2(-d'^2),
            // neg folds into the v_exp_f32 source modifier -> no scalar muls at all.
            As[e] = (_Float16)__builtin_amdgcn_exp2f(-ds);
            Ab[e] = (_Float16)__builtin_amdgcn_exp2f(-db);
        }

        // B = p^T chunk: lane holds class column n = lane%16, 16 contiguous j's
        const _Float16* pb = pT + (size_t)m16 * N_PIX + (j0 + jhalf);
        v16h B0 = *(const v16h*)pb;                          // classes 0..15
        v16h B1 = *(const v16h*)(pb + (size_t)16 * N_PIX);   // classes 16..31 (21=ones)

        accs0 = __builtin_amdgcn_wmma_f32_16x16x32_f16(false, As, false, B0, (short)0, accs0, false, false);
        accs1 = __builtin_amdgcn_wmma_f32_16x16x32_f16(false, As, false, B1, (short)0, accs1, false, false);
        accb0 = __builtin_amdgcn_wmma_f32_16x16x32_f16(false, Ab, false, B0, (short)0, accb0, false, false);
        accb1 = __builtin_amdgcn_wmma_f32_16x16x32_f16(false, Ab, false, B1, (short)0, accb1, false, false);
    }

    // spill accumulators (D layout: VGPR r -> row m = r + 8*hi, col n = lane%16)
    {
        const int mb = hi ? 8 : 0;
#pragma unroll
        for (int r = 0; r < 8; ++r) {
            MsL[w][mb + r][m16]      = accs0[r];
            MsL[w][mb + r][16 + m16] = accs1[r];
            MbL[w][mb + r][m16]      = accb0[r];
            MbL[w][mb + r][16 + m16] = accb1[r];
        }
    }
    __syncthreads();

    // reduce the 8 per-wave partials (only columns 0..21 matter; col 21 = row-sum norm)
    for (int idx = t; idx < 16 * 22; idx += WPB * 32) {
        int m = idx / 22, c = idx % 22;
        float s = 0.f, b = 0.f;
#pragma unroll
        for (int wv = 0; wv < WPB; ++wv) { s += MsL[wv][m][c]; b += MbL[wv][m][c]; }
        MsR[m][c] = s; MbR[m][c] = b;
    }
    __syncthreads();

    // finalize: q = u - (Ms/ns) @ Es - (Mb/nb) @ Eb
    for (int idx = t; idx < 16 * NCLS; idx += WPB * 32) {
        int m = idx / NCLS, c = idx % NCLS;
        int i = i0 + m;
        float inv_s = 1.0f / MsR[m][21];
        float inv_b = 1.0f / MbR[m][21];
        float as = 0.f, ab = 0.f;
#pragma unroll
        for (int cp = 0; cp < NCLS; ++cp) {
            as += MsR[m][cp] * es[cp * NCLS + c];
            ab += MbR[m][cp] * eb[cp * NCLS + c];
        }
        qout[(size_t)i * NCLS + c] = u[(size_t)i * NCLS + c] - as * inv_s - ab * inv_b;
    }
}

extern "C" void kernel_launch(void* const* d_in, const int* in_sizes, int n_in,
                              void* d_out, int out_size, void* d_ws, size_t ws_size,
                              hipStream_t stream) {
    (void)in_sizes; (void)n_in; (void)out_size; (void)ws_size;
    const float* u  = (const float*)d_in[0];  // unaries [1,80,80,21]
    const float* im = (const float*)d_in[1];  // image   [1,80,80,3]
    const float* sw = (const float*)d_in[2];  // spatial_w [21,21]
    const float* bw = (const float*)d_in[3];  // bilateral_w [21,21]
    const float* cm = (const float*)d_in[4];  // compat [21,21]

    char* ws = (char*)d_ws;                               // all offsets 512B-aligned
    float*    feat = (float*)(ws);                        // 6400*8*4   = 204800 B
    _Float16* pT   = (_Float16*)(ws + 204800);            // 32*6400*2  = 409600 B
    float*    qbuf = (float*)(ws + 614400);               // 6400*21*4  = 537600 B
    float*    es   = (float*)(ws + 1152000);              // 21*21*4
    float*    eb   = (float*)(ws + 1154048);              // 21*21*4
    float*    out  = (float*)d_out;

    crf_feat_kernel<<<(N_PIX + 255) / 256, 256, 0, stream>>>(im, feat);
    crf_mix_kernel<<<2, 256, 0, stream>>>(sw, bw, cm, es, eb);

    for (int it = 0; it < NITER; ++it) {
        const float* qin = (it == 0) ? u : qbuf;
        float* qo = (it == NITER - 1) ? out : qbuf;
        crf_softmax_kernel<<<(N_PIX + 255) / 256, 256, 0, stream>>>(qin, pT);
        crf_tile_kernel<<<N_PIX / 16, WPB * 32, 0, stream>>>(feat, pT, u, es, eb, qo);
    }
}
